// CrossAttentionFusion_46634754900418
// MI455X (gfx1250) — compile-verified, matrix operand order checked by reading, not run on hardware
//
#include <hip/hip_runtime.h>
#include <math.h>

#define DIM   384
#define HDIM  768
#define FDIM  1536
#define EPSV  1e-5f
#define MTILE 32

typedef __attribute__((ext_vector_type(16))) _Float16 v16h;
typedef __attribute__((ext_vector_type(8)))  _Float16 v8h;
typedef __attribute__((ext_vector_type(8)))  float    v8f;

// f16 weight cache layout in d_ws (element offsets)
#define OFF_WV_DM 0
#define OFF_WO_DM 147456
#define OFF_WV_MD 294912
#define OFF_WO_MD 442368
#define OFF_W1    589824
#define OFF_W2    1769472
#define W_TOTAL   2359296

#define LDS_BYTES 98304   // 96 KB dynamic LDS

// ---------------- weight f32 -> f16 conversion (runs once per launch) ----------------
__global__ void cvt_weights_kernel(const float* __restrict__ dm_in_w,
                                   const float* __restrict__ dm_out_w,
                                   const float* __restrict__ md_in_w,
                                   const float* __restrict__ md_out_w,
                                   const float* __restrict__ w1,
                                   const float* __restrict__ w2,
                                   _Float16* __restrict__ outw) {
  int i = blockIdx.x * 256 + threadIdx.x;
  if (i >= W_TOTAL) return;
  float v;
  if (i < OFF_WO_DM)      v = dm_in_w[2 * DIM * DIM + i];          // Wv rows of dm_in_w
  else if (i < OFF_WV_MD) v = dm_out_w[i - OFF_WO_DM];
  else if (i < OFF_WO_MD) v = md_in_w[2 * DIM * DIM + (i - OFF_WV_MD)];
  else if (i < OFF_W1)    v = md_out_w[i - OFF_WO_MD];
  else if (i < OFF_W2)    v = w1[i - OFF_W1];
  else                    v = w2[i - OFF_W2];
  outw[i] = (_Float16)v;
}

// CDNA5 16-bit A/B fragment: lane(r,h) chunk0 = K[h*8..h*8+7], chunk1 = K[16+h*8..+7]
__device__ __forceinline__ v16h ldfrag(const _Float16* p, int k0, int hh) {
  v8h lo = *(const v8h*)(p + k0 + hh * 8);
  v8h hi = *(const v8h*)(p + k0 + 16 + hh * 8);
  return __builtin_shufflevector(lo, hi, 0,1,2,3,4,5,6,7,8,9,10,11,12,13,14,15);
}

// NT column tiles x 2 row sub-tiles: A fragments reused across NT B tiles,
// each B fragment reused across 2 A sub-tiles (operand-reuse both ways).
template <int NT>
__device__ __forceinline__ void gemm_mt(const _Float16* arow0, const _Float16* arow1,
                                        const _Float16* wbase, size_t wstride,
                                        int K, int hh, v8f acc[][2]) {
  #pragma unroll 2
  for (int k0 = 0; k0 < K; k0 += 32) {
    v16h a0 = ldfrag(arow0, k0, hh);
    v16h a1 = ldfrag(arow1, k0, hh);
    #pragma unroll
    for (int j = 0; j < NT; ++j) {
      v16h b = ldfrag(wbase + (size_t)j * wstride, k0, hh);
      acc[j][0] = __builtin_amdgcn_wmma_f32_16x16x32_f16(false, a0, false, b, (short)0,
                                                         acc[j][0], false, false);
      acc[j][1] = __builtin_amdgcn_wmma_f32_16x16x32_f16(false, a1, false, b, (short)0,
                                                         acc[j][1], false, false);
    }
  }
}

// ---------------- fully fused per-32-row-tile kernel ----------------
__global__ __launch_bounds__(256)
void fused_kernel(const float* __restrict__ drug, const float* __restrict__ micro,
                  const float* __restrict__ dm_in_b, const float* __restrict__ dm_out_b,
                  const float* __restrict__ md_in_b, const float* __restrict__ md_out_b,
                  const float* __restrict__ nd_g, const float* __restrict__ nd_b,
                  const float* __restrict__ nm_g, const float* __restrict__ nm_b,
                  const float* __restrict__ fb1, const float* __restrict__ fb2,
                  const float* __restrict__ no_g, const float* __restrict__ no_b,
                  const _Float16* __restrict__ wf, float* __restrict__ out) {
  extern __shared__ __attribute__((aligned(16))) _Float16 lds[];   // 96 KB dynamic
  __shared__ float s_sum[32], s_sq[32], s_mu[32], s_rs[32];

  const int tid  = threadIdx.x;
  const int w    = tid >> 5;          // wave id 0..7
  const int lane = tid & 31;
  const int r    = lane & 15;         // A row within sub-tile / B,C,D column
  const int hh   = lane >> 4;
  const int m0   = blockIdx.x * MTILE;

  _Float16* bufM  = lds;              // 32x384 micro f16
  _Float16* bufD  = lds + 12288;      // 32x384 drug  f16
  _Float16* bufV1 = lds + 24576;      // 32x384 v_d
  _Float16* bufV2 = lds + 36864;      // 32x384 v_m
  _Float16* bufH  = lds;              // 32x768 concat LN outputs   (aliases bufM/bufD)
  _Float16* gbuf  = lds + 24576;      // 32x768 gelu chunk          (aliases bufV1/V2)

  const v8f vzero = {};

  // ---- stage input rows, cvt f32->f16 ----
  for (int i = tid; i < MTILE * DIM; i += 256) {
    bufM[i] = (_Float16)micro[(size_t)m0 * DIM + i];
    bufD[i] = (_Float16)drug[(size_t)m0 * DIM + i];
  }
  __syncthreads();

  // ---- V projections: v_d = micro@Wv_dm^T + b, v_m = drug@Wv_md^T + b ----
  for (int side = 0; side < 2; ++side) {
    const _Float16* A  = side ? bufD : bufM;
    const _Float16* Wv = wf + (side ? OFF_WV_MD : OFF_WV_DM);
    const float* bv    = (side ? md_in_b : dm_in_b) + 2 * DIM;
    _Float16* V        = side ? bufV2 : bufV1;

    v8f acc[3][2];
    #pragma unroll
    for (int j = 0; j < 3; ++j) { acc[j][0] = vzero; acc[j][1] = vzero; }
    gemm_mt<3>(A + r * DIM, A + (16 + r) * DIM,
               Wv + (size_t)(w * 16 + r) * DIM, (size_t)128 * DIM, DIM, hh, acc);
    #pragma unroll
    for (int j = 0; j < 3; ++j) {
      int col = (w + j * 8) * 16 + r;
      float bias = bv[col];
      #pragma unroll
      for (int s = 0; s < 2; ++s)
        #pragma unroll
        for (int v = 0; v < 8; ++v)
          V[(s * 16 + v + 8 * hh) * DIM + col] = (_Float16)(acc[j][s][v] + bias);
    }
  }
  __syncthreads();

  // ---- out-proj + residual + LayerNorm -> bufH (concat layout) ----
  for (int side = 0; side < 2; ++side) {
    const _Float16* V  = side ? bufV2 : bufV1;
    const _Float16* Wo = wf + (side ? OFF_WO_MD : OFF_WO_DM);
    const float* ob    = side ? md_out_b : dm_out_b;
    const float* res   = side ? micro : drug;
    const float* gA    = side ? nm_g : nd_g;
    const float* bA    = side ? nm_b : nd_b;
    const int cofs     = side ? DIM : 0;

    if (tid < 32) { s_sum[tid] = 0.f; s_sq[tid] = 0.f; }
    __syncthreads();

    v8f acc[3][2];
    #pragma unroll
    for (int j = 0; j < 3; ++j) { acc[j][0] = vzero; acc[j][1] = vzero; }
    gemm_mt<3>(V + r * DIM, V + (16 + r) * DIM,
               Wo + (size_t)(w * 16 + r) * DIM, (size_t)128 * DIM, DIM, hh, acc);

    float rr[3][2][8];
    #pragma unroll
    for (int j = 0; j < 3; ++j) {
      int col = (w + j * 8) * 16 + r;
      float bias = ob[col];
      #pragma unroll
      for (int s = 0; s < 2; ++s) {
        #pragma unroll
        for (int v = 0; v < 8; ++v) {
          int m = s * 16 + v + 8 * hh;
          float x = acc[j][s][v] + bias + res[(size_t)(m0 + m) * DIM + col];
          rr[j][s][v] = x;
          float sx = x, sq = x * x;
          #pragma unroll
          for (int d = 1; d < 16; d <<= 1) {
            sx += __shfl_xor(sx, d, 32);
            sq += __shfl_xor(sq, d, 32);
          }
          if (r == 0) { atomicAdd(&s_sum[m], sx); atomicAdd(&s_sq[m], sq); }
        }
      }
    }
    __syncthreads();
    if (tid < 32) {
      float mu  = s_sum[tid] * (1.0f / DIM);
      float var = s_sq[tid] * (1.0f / DIM) - mu * mu;
      s_mu[tid] = mu;
      s_rs[tid] = rsqrtf(var + EPSV);
    }
    __syncthreads();
    #pragma unroll
    for (int j = 0; j < 3; ++j) {
      int col = (w + j * 8) * 16 + r;
      float gg = gA[col], bo = bA[col];
      #pragma unroll
      for (int s = 0; s < 2; ++s)
        #pragma unroll
        for (int v = 0; v < 8; ++v) {
          int m = s * 16 + v + 8 * hh;
          bufH[m * HDIM + cofs + col] =
              (_Float16)((rr[j][s][v] - s_mu[m]) * s_rs[m] * gg + bo);
        }
    }
    __syncthreads();
  }

  // ---- FFN: 2 chunks of 768 hidden cols; FFN2 accumulates in registers ----
  v8f oacc[3][2];
  #pragma unroll
  for (int j = 0; j < 3; ++j) { oacc[j][0] = vzero; oacc[j][1] = vzero; }

  for (int chunk = 0; chunk < 2; ++chunk) {
    for (int grp = 0; grp < 2; ++grp) {           // FFN1: 6 col tiles/wave = 2 groups of 3
      int tn0 = w + grp * 24;
      v8f acc[3][2];
      #pragma unroll
      for (int j = 0; j < 3; ++j) { acc[j][0] = vzero; acc[j][1] = vzero; }
      gemm_mt<3>(bufH + r * HDIM, bufH + (16 + r) * HDIM,
                 wf + OFF_W1 + (size_t)(chunk * HDIM + tn0 * 16 + r) * HDIM,
                 (size_t)128 * HDIM, HDIM, hh, acc);
      #pragma unroll
      for (int j = 0; j < 3; ++j) {
        int tn   = tn0 + j * 8;
        int ncol = chunk * HDIM + tn * 16 + r;
        float bias = fb1[ncol];
        #pragma unroll
        for (int s = 0; s < 2; ++s)
          #pragma unroll
          for (int v = 0; v < 8; ++v) {
            float x = acc[j][s][v] + bias;
            float y = 0.5f * x * (1.0f + erff(x * 0.70710678118654752f));  // exact gelu
            gbuf[(s * 16 + v + 8 * hh) * HDIM + tn * 16 + r] = (_Float16)y;
          }
      }
    }
    __syncthreads();
    gemm_mt<3>(gbuf + r * HDIM, gbuf + (16 + r) * HDIM,            // FFN2 partial
               wf + OFF_W2 + (size_t)(w * 16 + r) * FDIM + chunk * HDIM,
               (size_t)128 * FDIM, HDIM, hh, oacc);
    __syncthreads();
  }

  // ---- final bias + LayerNorm -> f32 output ----
  if (tid < 32) { s_sum[tid] = 0.f; s_sq[tid] = 0.f; }
  __syncthreads();
  float rr2[3][2][8];
  #pragma unroll
  for (int j = 0; j < 3; ++j) {
    int col = (w + j * 8) * 16 + r;
    float bias = fb2[col];
    #pragma unroll
    for (int s = 0; s < 2; ++s) {
      #pragma unroll
      for (int v = 0; v < 8; ++v) {
        int m = s * 16 + v + 8 * hh;
        float x = oacc[j][s][v] + bias;
        rr2[j][s][v] = x;
        float sx = x, sq = x * x;
        #pragma unroll
        for (int d = 1; d < 16; d <<= 1) {
          sx += __shfl_xor(sx, d, 32);
          sq += __shfl_xor(sq, d, 32);
        }
        if (r == 0) { atomicAdd(&s_sum[m], sx); atomicAdd(&s_sq[m], sq); }
      }
    }
  }
  __syncthreads();
  if (tid < 32) {
    float mu  = s_sum[tid] * (1.0f / DIM);
    float var = s_sq[tid] * (1.0f / DIM) - mu * mu;
    s_mu[tid] = mu;
    s_rs[tid] = rsqrtf(var + EPSV);
  }
  __syncthreads();
  #pragma unroll
  for (int j = 0; j < 3; ++j) {
    int col = (w + j * 8) * 16 + r;
    float gg = no_g[col], bo = no_b[col];
    #pragma unroll
    for (int s = 0; s < 2; ++s)
      #pragma unroll
      for (int v = 0; v < 8; ++v) {
        int m = s * 16 + v + 8 * hh;
        out[(size_t)(m0 + m) * DIM + col] = (rr2[j][s][v] - s_mu[m]) * s_rs[m] * gg + bo;
      }
  }
}

extern "C" void kernel_launch(void* const* d_in, const int* in_sizes, int n_in,
                              void* d_out, int out_size, void* d_ws, size_t ws_size,
                              hipStream_t stream) {
  const float* drug     = (const float*)d_in[0];
  const float* micro    = (const float*)d_in[1];
  const float* dm_in_w  = (const float*)d_in[2];
  const float* dm_in_b  = (const float*)d_in[3];
  const float* dm_out_w = (const float*)d_in[4];
  const float* dm_out_b = (const float*)d_in[5];
  const float* md_in_w  = (const float*)d_in[6];
  const float* md_in_b  = (const float*)d_in[7];
  const float* md_out_w = (const float*)d_in[8];
  const float* md_out_b = (const float*)d_in[9];
  const float* nd_g     = (const float*)d_in[10];
  const float* nd_b     = (const float*)d_in[11];
  const float* nm_g     = (const float*)d_in[12];
  const float* nm_b     = (const float*)d_in[13];
  const float* w1       = (const float*)d_in[14];
  const float* b1       = (const float*)d_in[15];
  const float* w2       = (const float*)d_in[16];
  const float* b2       = (const float*)d_in[17];
  const float* no_g     = (const float*)d_in[18];
  const float* no_b     = (const float*)d_in[19];
  float* out            = (float*)d_out;
  _Float16* wf          = (_Float16*)d_ws;    // 4.72 MB f16 weight cache

  int B = in_sizes[0] / DIM;                  // 65536

  (void)hipFuncSetAttribute(reinterpret_cast<const void*>(&fused_kernel),
                            hipFuncAttributeMaxDynamicSharedMemorySize, LDS_BYTES);

  cvt_weights_kernel<<<(W_TOTAL + 255) / 256, 256, 0, stream>>>(
      dm_in_w, dm_out_w, md_in_w, md_out_w, w1, w2, wf);

  fused_kernel<<<B / MTILE, 256, LDS_BYTES, stream>>>(
      drug, micro, dm_in_b, dm_out_b, md_in_b, md_out_b,
      nd_g, nd_b, nm_g, nm_b, b1, b2, no_g, no_b, wf, out);
}
